// MLP_21835613733211
// MI455X (gfx1250) — matrix operand order repaired and checked by reading, not compiled, not run on hardware
//
#include <hip/hip_runtime.h>
#include <math.h>

typedef __attribute__((ext_vector_type(16))) _Float16 v16h;
typedef __attribute__((ext_vector_type(8)))  float    v8f;

#define BATCH 4096
#define SEQ   200
#define EMB   512
#define PF    8          // prefetch distance (rows ahead) in the gather loop

// ---------------------------------------------------------------------------
// Stage 0: convert fp32 weight [K,N] -> f16 transposed [N,K] (so WMMA B-frags
// are contiguous b128 loads along K).
// ---------------------------------------------------------------------------
__global__ void wt_to_f16_t(const float* __restrict__ W, _Float16* __restrict__ Wt,
                            int K, int N) {
    int tid = blockIdx.x * blockDim.x + threadIdx.x;
    if (tid >= K * N) return;
    int k = tid / N, n = tid % N;
    Wt[(size_t)n * K + k] = (_Float16)W[tid];
}

// ---------------------------------------------------------------------------
// Stage 1: embedding-bag mean.  One block per batch row, indices cached in
// LDS, each thread accumulates 2 of the 512 columns (fully coalesced 2KB row
// reads), software prefetch PF rows ahead, writes packed f16x2.
// ---------------------------------------------------------------------------
__global__ void embed_mean_f16(const int* __restrict__ idx, const float* __restrict__ E,
                               _Float16* __restrict__ sent) {
    __shared__ int sidx[SEQ];
    const int row = blockIdx.x;
    for (int s = threadIdx.x; s < SEQ; s += blockDim.x)
        sidx[s] = idx[row * SEQ + s];
    __syncthreads();

    const int col = threadIdx.x * 2;             // 256 thr * 2 = 512
    float a0 = 0.f, a1 = 0.f;
    for (int s = 0; s < SEQ; ++s) {
        if (s + PF < SEQ)
            __builtin_prefetch(E + (size_t)sidx[s + PF] * EMB + col, 0, 1);
        const float* erow = E + (size_t)sidx[s] * EMB + col;
        a0 += erow[0];
        a1 += erow[1];
    }
    const float inv = 1.0f / (float)SEQ;
    union { _Float16 h[2]; unsigned u; } p;
    p.h[0] = (_Float16)(a0 * inv);
    p.h[1] = (_Float16)(a1 * inv);
    *reinterpret_cast<unsigned*>(sent + (size_t)row * EMB + col) = p.u;
}

// ---------------------------------------------------------------------------
// Stage 2: C[M,N] = relu(A[M,K] @ W[K,N] + bias), f16 in / f32 acc / f16 out.
// One wave32 per 16x64 tile of C: one A fragment feeds 4 independent WMMAs
// per K-step (4x less A traffic, 4-deep WMMA pipelining).  Fragment
// addressing puts K contiguous per lane half (k = 16h + s) in both A and B;
// all loads are 16B-aligned b128s.  Requires N % 64 == 0.
// ---------------------------------------------------------------------------
__global__ void gemm_bias_relu_wmma(const _Float16* __restrict__ A,   // [M,K]
                                    const _Float16* __restrict__ Bt,  // [N,K]
                                    const float* __restrict__ bias,   // [N]
                                    _Float16* __restrict__ C,         // [M,N]
                                    int M, int N, int K) {
    const int gwave = (blockIdx.x * blockDim.x + threadIdx.x) >> 5;
    const int lane  = threadIdx.x & 31;
    const int ntile = N >> 6;                    // 64-wide column tiles
    const int tm = (gwave / ntile) << 4;
    const int tn = (gwave % ntile) << 6;
    const int mn   = lane & 15;                  // A row / B col / C col (mod 16)
    const int half = lane >> 4;

    const _Float16* arow = A + (size_t)(tm + mn) * K;
    const _Float16* brow[4];
#pragma unroll
    for (int j = 0; j < 4; ++j)
        brow[j] = Bt + (size_t)(tn + j * 16 + mn) * K;

    v8f c0 = {}, c1 = {}, c2 = {}, c3 = {};
    for (int k0 = 0; k0 < K; k0 += 32) {
        union { v16h v; uint4 q[2]; } a, b0, b1, b2, b3;
        // A lane: K = [16h..16h+15] contiguous
        a.q[0] = *reinterpret_cast<const uint4*>(arow + k0 + half * 16);
        a.q[1] = *reinterpret_cast<const uint4*>(arow + k0 + half * 16 + 8);
        // B lane: K = [16h..16h+15] contiguous
        const int boff = k0 + half * 16;
        b0.q[0] = *reinterpret_cast<const uint4*>(brow[0] + boff);
        b0.q[1] = *reinterpret_cast<const uint4*>(brow[0] + boff + 8);
        b1.q[0] = *reinterpret_cast<const uint4*>(brow[1] + boff);
        b1.q[1] = *reinterpret_cast<const uint4*>(brow[1] + boff + 8);
        b2.q[0] = *reinterpret_cast<const uint4*>(brow[2] + boff);
        b2.q[1] = *reinterpret_cast<const uint4*>(brow[2] + boff + 8);
        b3.q[0] = *reinterpret_cast<const uint4*>(brow[3] + boff);
        b3.q[1] = *reinterpret_cast<const uint4*>(brow[3] + boff + 8);

        c0 = __builtin_amdgcn_wmma_f32_16x16x32_f16(false, a.v, false, b0.v,
                                                    (short)0, c0, false, false);
        c1 = __builtin_amdgcn_wmma_f32_16x16x32_f16(false, a.v, false, b1.v,
                                                    (short)0, c1, false, false);
        c2 = __builtin_amdgcn_wmma_f32_16x16x32_f16(false, a.v, false, b2.v,
                                                    (short)0, c2, false, false);
        c3 = __builtin_amdgcn_wmma_f32_16x16x32_f16(false, a.v, false, b3.v,
                                                    (short)0, c3, false, false);
    }

#pragma unroll
    for (int j = 0; j < 4; ++j) {
        const v8f& c = (j == 0) ? c0 : (j == 1) ? c1 : (j == 2) ? c2 : c3;
        const int ncol = tn + j * 16 + mn;
        const float bv = bias[ncol];
#pragma unroll
        for (int r = 0; r < 8; ++r) {
            const int m = (half << 3) + r;       // C row for accumulator r
            float v = c[r] + bv;
            v = v > 0.f ? v : 0.f;
            C[(size_t)(tm + m) * N + ncol] = (_Float16)v;
        }
    }
}

// ---------------------------------------------------------------------------
// Stage 3: head (64->3) + softmax, fp32 out.  One thread per batch row.
// ---------------------------------------------------------------------------
__global__ void head_softmax(const _Float16* __restrict__ H,  // [BATCH,64]
                             const float* __restrict__ W5,    // [64,3]
                             const float* __restrict__ b5,    // [3]
                             float* __restrict__ out) {       // [BATCH,3]
    const int row = blockIdx.x * blockDim.x + threadIdx.x;
    if (row >= BATCH) return;
    const _Float16* h = H + (size_t)row * 64;
    float a0 = b5[0], a1 = b5[1], a2 = b5[2];
#pragma unroll
    for (int k = 0; k < 64; ++k) {
        const float x = (float)h[k];
        a0 += x * W5[k * 3 + 0];
        a1 += x * W5[k * 3 + 1];
        a2 += x * W5[k * 3 + 2];
    }
    const float mx = fmaxf(a0, fmaxf(a1, a2));
    const float e0 = __expf(a0 - mx), e1 = __expf(a1 - mx), e2 = __expf(a2 - mx);
    const float inv = 1.0f / (e0 + e1 + e2);
    out[row * 3 + 0] = e0 * inv;
    out[row * 3 + 1] = e1 * inv;
    out[row * 3 + 2] = e2 * inv;
}

// ---------------------------------------------------------------------------
// Workspace layout (bytes)
// ---------------------------------------------------------------------------
#define O_SENT 0u                                   // 4096*512*2 = 4 MiB
#define O_H1   (O_SENT + BATCH * 512u * 2u)
#define O_H2   (O_H1   + BATCH * 512u * 2u)
#define O_H3   (O_H2   + BATCH * 256u * 2u)
#define O_H4   (O_H3   + BATCH * 128u * 2u)
#define O_W1T  (O_H4   + BATCH *  64u * 2u)
#define O_W2T  (O_W1T  + 512u * 512u * 2u)
#define O_W3T  (O_W2T  + 512u * 256u * 2u)
#define O_W4T  (O_W3T  + 256u * 128u * 2u)

extern "C" void kernel_launch(void* const* d_in, const int* in_sizes, int n_in,
                              void* d_out, int out_size, void* d_ws, size_t ws_size,
                              hipStream_t stream) {
    const int*   idx = (const int*)  d_in[0];
    const float* E   = (const float*)d_in[1];
    const float* W1  = (const float*)d_in[2];  const float* b1 = (const float*)d_in[3];
    const float* W2  = (const float*)d_in[4];  const float* b2 = (const float*)d_in[5];
    const float* W3  = (const float*)d_in[6];  const float* b3 = (const float*)d_in[7];
    const float* W4  = (const float*)d_in[8];  const float* b4 = (const float*)d_in[9];
    const float* W5  = (const float*)d_in[10]; const float* b5 = (const float*)d_in[11];
    float* out = (float*)d_out;

    char* ws = (char*)d_ws;
    _Float16* sent = (_Float16*)(ws + O_SENT);
    _Float16* h1   = (_Float16*)(ws + O_H1);
    _Float16* h2   = (_Float16*)(ws + O_H2);
    _Float16* h3   = (_Float16*)(ws + O_H3);
    _Float16* h4   = (_Float16*)(ws + O_H4);
    _Float16* w1t  = (_Float16*)(ws + O_W1T);
    _Float16* w2t  = (_Float16*)(ws + O_W2T);
    _Float16* w3t  = (_Float16*)(ws + O_W3T);
    _Float16* w4t  = (_Float16*)(ws + O_W4T);

    // Stage 0: weight convert+transpose (tiny: < 1 MB total)
    wt_to_f16_t<<<(512 * 512 + 255) / 256, 256, 0, stream>>>(W1, w1t, 512, 512);
    wt_to_f16_t<<<(512 * 256 + 255) / 256, 256, 0, stream>>>(W2, w2t, 512, 256);
    wt_to_f16_t<<<(256 * 128 + 255) / 256, 256, 0, stream>>>(W3, w3t, 256, 128);
    wt_to_f16_t<<<(128 *  64 + 255) / 256, 256, 0, stream>>>(W4, w4t, 128,  64);

    // Stage 1: embedding bag (bandwidth-bound stage)
    embed_mean_f16<<<BATCH, 256, 0, stream>>>(idx, E, sent);

    // Stage 2: WMMA MLP. blocks = (M/16)*(N/64) tiles / 8 waves-per-block
    gemm_bias_relu_wmma<<<(BATCH / 16) * (512 / 64) / 8, 256, 0, stream>>>(sent, w1t, b1, h1, BATCH, 512, 512);
    gemm_bias_relu_wmma<<<(BATCH / 16) * (256 / 64) / 8, 256, 0, stream>>>(h1,   w2t, b2, h2, BATCH, 256, 512);
    gemm_bias_relu_wmma<<<(BATCH / 16) * (128 / 64) / 8, 256, 0, stream>>>(h2,   w3t, b3, h3, BATCH, 128, 256);
    gemm_bias_relu_wmma<<<(BATCH / 16) * ( 64 / 64) / 8, 256, 0, stream>>>(h3,   w4t, b4, h4, BATCH,  64, 128);

    // Stage 3: head + softmax
    head_softmax<<<BATCH / 256, 256, 0, stream>>>(h4, W5, b5, out);
}
